// ScaleDotProductAttention_52991306498190
// MI455X (gfx1250) — matrix operand order repaired and checked by reading, not compiled, not run on hardware
//
#include <hip/hip_runtime.h>

typedef __bf16 bf16_t;
typedef __attribute__((ext_vector_type(16))) __bf16 v16bf;
typedef __attribute__((ext_vector_type(8)))  float  v8f;
typedef __attribute__((ext_vector_type(4)))  float  v4f;
typedef __attribute__((ext_vector_type(4)))  unsigned short us4;

#define NB 2
#define NH 16
#define SEQ 2048
#define DH 64
// 1/sqrt(64) * log2(e): scores come out of QK^T already in log2 domain
#define QSCALE 0.18033688011112042f
#define MASKVAL (-1.0e30f)   // finite "-inf": exp2(MASKVAL - anything) == 0

#define KSTR 68   // LDS row stride (halfwords) for K tiles : conflict-free
#define VSTR 36   // LDS row stride (halfwords) for V^T     : conflict-free
#define PSTR 34   // LDS row stride (halfwords) for P       : conflict-free

union AFrag { v16bf v; bf16_t e[16]; unsigned short u[16]; unsigned int w[8]; };
union CFrag { v8f  v; float  e[8]; };

__device__ __forceinline__ unsigned short bfbits(float x) {
  return __builtin_bit_cast(unsigned short, (bf16_t)x);
}

// A-matrix (16x32 bf16) per-lane packed-pair K index (always even)
__device__ __forceinline__ int a_pair_base(int vv, int half) {
  return (vv < 4 ? 2 * vv : 16 + 2 * (vv - 4)) + half * 8;
}

__global__ __launch_bounds__(256, 1) void fa_fwd_kernel(
    const float* __restrict__ qg, const float* __restrict__ kg,
    const float* __restrict__ vg, float* __restrict__ outg,
    float* __restrict__ attng)
{
  // Cooperative staging buffers (one (b,h) per block, shared by 8 waves)
  __shared__ unsigned short khi[32][KSTR];        // K tile(s), bf16 high part
  __shared__ unsigned short klo[32][KSTR];        // K tile(s), bf16 residual
  __shared__ unsigned short vT [DH][VSTR];        // V chunk, transposed (d-major)
  __shared__ unsigned short pbuf[8][16][PSTR];    // per-wave P staging (16x32)

  const int lane = threadIdx.x & 31;
  const int wave = threadIdx.x >> 5;
  const int bh   = blockIdx.x >> 4;   // 0..31  (b*H+h)
  const int qblk = blockIdx.x & 15;   // 16 blocks of 128 query rows
  const int q0   = qblk * 128 + wave * 16;
  const int half = lane >> 4;
  const int l16  = lane & 15;
  const int dtri = l16 - 8 * half;    // diagonal-tile mask: masked iff dtri > vc

  const float* qb = qg + ((size_t)bh * SEQ + q0) * DH;
  const float* kb = kg + (size_t)bh * SEQ * DH;
  const float* vb = vg + (size_t)bh * SEQ * DH;
  float* ob = outg  + ((size_t)bh * SEQ + q0) * DH;
  float* ab = attng + (size_t)bh * SEQ * SEQ;

  const int ktq    = q0 >> 4;           // this wave's diagonal key-tile
  const int ktq_wg = qblk * 8 + 7;      // block-wide last key-tile (uniform)

  // ---- cooperative staging helpers (all 256 threads) ----
  auto stage_k = [&](int kt, int rbase) {
    const int t   = threadIdx.x;
    const int key = t >> 4;             // 0..15
    const int d4  = (t & 15) << 2;      // 0..60
    const float* p = kb + (size_t)(kt * 16 + key) * DH + d4;
    float x0 = p[0], x1 = p[1], x2 = p[2], x3 = p[3];
    bf16_t h0 = (bf16_t)x0, h1 = (bf16_t)x1, h2 = (bf16_t)x2, h3 = (bf16_t)x3;
    us4 hv = { __builtin_bit_cast(unsigned short, h0),
               __builtin_bit_cast(unsigned short, h1),
               __builtin_bit_cast(unsigned short, h2),
               __builtin_bit_cast(unsigned short, h3) };
    *(us4*)&khi[rbase + key][d4] = hv;
    us4 lv = { bfbits(x0 - (float)h0), bfbits(x1 - (float)h1),
               bfbits(x2 - (float)h2), bfbits(x3 - (float)h3) };
    *(us4*)&klo[rbase + key][d4] = lv;
  };
  auto stage_v = [&](int kb0) {         // 32 keys x 64 d, transposed into vT
    const int t   = threadIdx.x;
    const int key = t >> 3;             // 0..31
    const int d8  = (t & 7) << 3;       // 0..56
    const float* p = vb + (size_t)(kb0 + key) * DH + d8;
#pragma unroll
    for (int j = 0; j < 8; ++j)
      vT[d8 + j][key] = bfbits(p[j]);
  };

  // ---- Q fragments: pre-scaled by 0.125*log2e, bf16 hi/lo split ----
  AFrag qhi[2], qlo[2];
#pragma unroll
  for (int ch = 0; ch < 2; ++ch) {
#pragma unroll
    for (int vv = 0; vv < 8; ++vv) {
      int d0 = ch * 32 + a_pair_base(vv, half);
      const float* p = qb + (size_t)l16 * DH + d0;
      float x = p[0] * QSCALE, y = p[1] * QSCALE;
      bf16_t xh = (bf16_t)x, yh = (bf16_t)y;
      qhi[ch].e[2*vv]   = xh;  qhi[ch].e[2*vv+1] = yh;
      qlo[ch].e[2*vv]   = (bf16_t)(x - (float)xh);
      qlo[ch].e[2*vv+1] = (bf16_t)(y - (float)yh);
    }
  }

  // S-tile (log2 domain) from LDS-staged K: 6 bf16 WMMAs (hi/lo correction)
  auto compute_S = [&](int rbase, CFrag& c) {
#pragma unroll
    for (int i = 0; i < 8; ++i) c.e[i] = 0.0f;
#pragma unroll
    for (int ch = 0; ch < 2; ++ch) {
      AFrag khf, klf;                   // packed b32 loads, zero-ALU assembly
#pragma unroll
      for (int vv = 0; vv < 8; ++vv) {
        int col = ch * 32 + a_pair_base(vv, half);
        khf.w[vv] = *(const unsigned int*)&khi[rbase + l16][col];
        klf.w[vv] = *(const unsigned int*)&klo[rbase + l16][col];
      }
      c.v = __builtin_amdgcn_wmma_f32_16x16x32_bf16(false, qhi[ch].v, false, khf.v, (short)0, c.v, false, false);
      c.v = __builtin_amdgcn_wmma_f32_16x16x32_bf16(false, qlo[ch].v, false, khf.v, (short)0, c.v, false, false);
      c.v = __builtin_amdgcn_wmma_f32_16x16x32_bf16(false, qhi[ch].v, false, klf.v, (short)0, c.v, false, false);
    }
  };

  // ---- Pass 1: per-lane online (m, l); no cross-lane traffic per tile ----
  float mrun[8], lrun[8];
#pragma unroll
  for (int i = 0; i < 8; ++i) { mrun[i] = MASKVAL; lrun[i] = 0.0f; }

  for (int kt = 0; kt <= ktq_wg; ++kt) {
    __syncthreads();                    // protect previous reads
    stage_k(kt, 0);
    if (kt < ktq_wg)
      __builtin_prefetch(kb + (size_t)((kt + 1) * 16 + (threadIdx.x >> 4)) * DH, 0, 1);
    __syncthreads();                    // staging visible
    if (kt <= ktq) {
      CFrag c;
      compute_S(0, c);
      if (kt == ktq) {                  // only the diagonal tile needs masking
#pragma unroll
        for (int vc = 0; vc < 8; ++vc)
          if (dtri > vc) c.e[vc] = MASKVAL;
      }
#pragma unroll
      for (int vc = 0; vc < 8; ++vc) {
        float s = c.e[vc];
        float mnew = fmaxf(mrun[vc], s);
        lrun[vc] = lrun[vc] * __builtin_exp2f(mrun[vc] - mnew)
                 + __builtin_exp2f(s - mnew);
        mrun[vc] = mnew;
      }
    }
  }
  // merge (m,l) partials across the 16 lanes sharing each row (once per wave)
#pragma unroll
  for (int off = 1; off < 16; off <<= 1) {
#pragma unroll
    for (int vc = 0; vc < 8; ++vc) {
      float mo = __shfl_xor(mrun[vc], off, 32);
      float lo = __shfl_xor(lrun[vc], off, 32);
      float mn = fmaxf(mrun[vc], mo);
      lrun[vc] = lrun[vc] * __builtin_exp2f(mrun[vc] - mn)
               + lo       * __builtin_exp2f(mo - mn);
      mrun[vc] = mn;
    }
  }
  float invl[8];
#pragma unroll
  for (int i = 0; i < 8; ++i) invl[i] = 1.0f / lrun[i];

  // ---- Pass 2: recompute S, emit normalized P, accumulate O = P.V ----
  CFrag accu[4];
#pragma unroll
  for (int ct = 0; ct < 4; ++ct)
#pragma unroll
    for (int i = 0; i < 8; ++i) accu[ct].e[i] = 0.0f;

  const int nchw = (ktq_wg + 2) >> 1;   // uniform chunk count (ktq_wg is odd)
  for (int kc = 0; kc < nchw; ++kc) {
    __syncthreads();                    // protect previous reads
    stage_k(kc * 2,     0);
    stage_k(kc * 2 + 1, 16);
    stage_v(kc * 32);
    __syncthreads();                    // staging visible

    if (kc * 2 <= ktq) {
#pragma unroll
      for (int sub = 0; sub < 2; ++sub) {
        int kt = kc * 2 + sub;
        if (kt <= ktq) {
          CFrag c;
          compute_S(sub * 16, c);
          if (kt == ktq) {
#pragma unroll
            for (int vc = 0; vc < 8; ++vc)
              if (dtri > vc) c.e[vc] = MASKVAL;
          }
#pragma unroll
          for (int vc = 0; vc < 8; ++vc) {
            int m = vc + 8 * half;
            int key = kt * 16 + l16;
            float p = __builtin_exp2f(c.e[vc] - mrun[vc]) * invl[vc];
            __builtin_nontemporal_store(p, ab + (size_t)(q0 + m) * SEQ + key);
            pbuf[wave][m][sub * 16 + l16] =
                __builtin_bit_cast(unsigned short, (bf16_t)p);
          }
        } else {
          // half-chunk past the diagonal: P = 0 (attn zeros come from tail fill)
#pragma unroll
          for (int r = 0; r < 8; ++r) {
            int flat = r * 32 + lane;
            pbuf[wave][flat >> 4][sub * 16 + (flat & 15)] = 0;
          }
        }
      }
      asm volatile("s_wait_dscnt 0" ::: "memory");   // wave-private LDS RAW

      // reload P as A-matrix fragment (16x32 bf16), packed b32 loads
      AFrag pa;
#pragma unroll
      for (int vv = 0; vv < 8; ++vv) {
        int k0 = a_pair_base(vv, half);
        pa.w[vv] = *(const unsigned int*)&pbuf[wave][l16][k0];
      }

#pragma unroll
      for (int ct = 0; ct < 4; ++ct) {   // 4 column tiles of 16 (D=64)
        AFrag vf;                        // B-matrix 32x16 from transposed LDS V
#pragma unroll
        for (int vv = 0; vv < 8; ++vv) {
          int row = ct * 16 + l16;       // d index
          int col = a_pair_base(vv, half);  // key pair (table order)
          vf.w[vv] = *(const unsigned int*)&vT[row][col];
        }
        accu[ct].v = __builtin_amdgcn_wmma_f32_16x16x32_bf16(
            false, pa.v, false, vf.v, (short)0, accu[ct].v, false, false);
      }
    }
  }

  // ---- store O ----
#pragma unroll
  for (int ct = 0; ct < 4; ++ct)
#pragma unroll
    for (int vc = 0; vc < 8; ++vc) {
      int m = vc + 8 * half;
      ob[(size_t)m * DH + ct * 16 + l16] = accu[ct].e[vc];
    }

  // ---- zero-fill strictly-future attention columns (streaming NT stores) ----
  const int z0  = (ktq + 1) * 16;
  const int cnt = SEQ - z0;
  if (cnt > 0) {
    v4f zz = {0.0f, 0.0f, 0.0f, 0.0f};
    for (int m = 0; m < 16; ++m) {
      float* row = ab + (size_t)(q0 + m) * SEQ + z0;
      for (int kk = lane * 4; kk < cnt; kk += 128)
        __builtin_nontemporal_store(zz, (v4f*)(row + kk));
    }
  }
}

extern "C" void kernel_launch(void* const* d_in, const int* in_sizes, int n_in,
                              void* d_out, int out_size, void* d_ws, size_t ws_size,
                              hipStream_t stream) {
  (void)in_sizes; (void)n_in; (void)out_size; (void)d_ws; (void)ws_size;
  const float* q = (const float*)d_in[0];
  const float* k = (const float*)d_in[1];
  const float* v = (const float*)d_in[2];
  // d_in[3] is the tril mask; causality is synthesized in-kernel.
  float* out  = (float*)d_out;
  float* attn = out + (size_t)NB * NH * SEQ * DH;   // tuple: (output, attention)
  dim3 grid(NB * NH * (SEQ / 128));   // 512 workgroups
  dim3 block(256);                    // 8 wave32 per WG
  hipLaunchKernelGGL(fa_fwd_kernel, grid, block, 0, stream, q, k, v, out, attn);
}
